// TFAM_21217138442590
// MI455X (gfx1250) — matrix operand order repaired and checked by reading, not compiled, not run on hardware
//
#include <hip/hip_runtime.h>
#include <hip/hip_bf16.h>
#include <stdint.h>

typedef __attribute__((ext_vector_type(16))) __bf16        v16bf;
typedef __attribute__((ext_vector_type(8)))  float         v8f;
typedef __attribute__((ext_vector_type(8)))  unsigned int  v8u;
typedef __attribute__((ext_vector_type(4)))  unsigned int  u32x4;

#define BB 4
#define CC 256
#define NN 4096

// Hardware bf16 convert (v_cvt_pk_bf16_f32 on gfx1250) instead of soft rounding
__device__ __forceinline__ unsigned short f2bf(float f) {
    __bf16 h = (__bf16)f;
    return __builtin_bit_cast(unsigned short, h);
}

// Register image of one WMMA A/B operand (16 bf16 = 8 VGPRs)
union ABReg {
    v16bf          v;
    unsigned short us[16];
    u32x4          q[2];   // two 16B chunks (A-operand pattern)
    v8u            w;      // one 32B chunk (B-operand pattern)
};

// Async global->LDS 16B copy, tracked by ASYNCcnt (CDNA5).
// LDS address = low 32 bits of the generic pointer (ISA 10.2 aperture rule).
__device__ __forceinline__ void async_copy_b128(void* lds_dst, const void* gsrc) {
    unsigned lds = (unsigned)(size_t)lds_dst;
    asm volatile("global_load_async_to_lds_b128 %0, %1, off"
                 :: "v"(lds), "v"(gsrc) : "memory");
}
__device__ __forceinline__ void wait_async0() {
#if __has_builtin(__builtin_amdgcn_s_wait_asynccnt)
    __builtin_amdgcn_s_wait_asynccnt(0);
#else
    asm volatile("s_wait_asynccnt 0x0" ::: "memory");
#endif
}

// ---------------------------------------------------------------------------
// Phase 1: 1x1-conv projections.  Y^T[b][n][o] = sum_c W[o][c]*X[b][c][n] + bias[o]
// Stored bf16 [B,N,C].  The proj==1 pass additionally writes Vt = bf16(x2) in
// [B,C,N] layout (data is already staged/converted for the K projection).
// ---------------------------------------------------------------------------
__global__ __launch_bounds__(256) void proj_kernel(
    const float* __restrict__ x1, const float* __restrict__ x2,
    const float* __restrict__ w1, const float* __restrict__ b1,
    const float* __restrict__ w2, const float* __restrict__ b2,
    unsigned short* __restrict__ Qt, unsigned short* __restrict__ Kt,
    unsigned short* __restrict__ Vt)
{
    __shared__ unsigned short Xs[16][CC];   // X^T tile: [token][channel], 8 KB

    const int pid  = blockIdx.x;
    const int proj = pid & 1;
    const int bt   = pid >> 1;
    const int b    = bt / (NN / 16);
    const int n0   = (bt % (NN / 16)) * 16;

    const float* X    = proj ? x2 : x1;
    const float* W    = proj ? w2 : w1;
    const float* bias = proj ? b2 : b1;
    unsigned short* Y = proj ? Kt : Qt;

    const int t = threadIdx.x;
    {   // cooperative transpose-stage: Xs[n][c] = bf16(X[b][c][n0+n])
        const int tn = t & 15;
        for (int c = t >> 4; c < CC; c += 16) {
            const unsigned short v = f2bf(X[((size_t)b * CC + c) * NN + n0 + tn]);
            Xs[tn][c] = v;
            if (proj) Vt[((size_t)b * CC + c) * NN + n0 + tn] = v;  // bf16 V copy
        }
    }
    __syncthreads();

    const int w    = t >> 5;
    const int lane = t & 31;
    const int m16  = lane & 15;
    const int h    = lane >> 4;

    #pragma unroll
    for (int tile = 0; tile < 2; ++tile) {
        const int o0 = w * 32 + tile * 16;
        const int oc = o0 + m16;              // this lane's output channel (B-col)
        v8f acc;
        const float bv = bias[oc];
        #pragma unroll
        for (int r = 0; r < 8; ++r) acc[r] = bv;

        #pragma unroll
        for (int kc = 0; kc < CC; kc += 32) {
            ABReg a, bmat;
            // A (16 tokens x 32 c): lane m=L%16, K halves at h*8 and 16+h*8
            a.q[0] = *(const u32x4*)&Xs[m16][kc + h * 8];
            a.q[1] = *(const u32x4*)&Xs[m16][kc + 16 + h * 8];
            // B (32 c x 16 o): lane n=L%16, K halves at h*8 and 16+h*8
            const float* wrow = &W[(size_t)oc * CC + kc];
            #pragma unroll
            for (int e = 0; e < 16; ++e) bmat.us[e] = f2bf(wrow[e < 8 ? h * 8 + e : 16 + h * 8 + (e - 8)]);
            acc = __builtin_amdgcn_wmma_f32_16x16x32_bf16(
                      false, a.v, false, bmat.v, (short)0, acc, false, false);
        }
        // D layout: VGPR r -> row m = r + 8h, col n = m16
        #pragma unroll
        for (int r = 0; r < 8; ++r) {
            const int row = r + 8 * h;
            Y[((size_t)b * NN + n0 + row) * CC + oc] = f2bf(acc[r]);
        }
    }
}

// ---------------------------------------------------------------------------
// Phase 2: fused flash attention.
//   S = Qt . Kt^T (bf16 WMMA, f32 acc), online softmax over 128-key blocks,
//   O^T[i][c] += P . V^T.  V block staged via async global->LDS b128 copies
//   (ASYNCcnt), overlapped with the S matmul + softmax, drained before P.V.
// ---------------------------------------------------------------------------
__global__ __launch_bounds__(256) void attn_kernel(
    const unsigned short* __restrict__ Qt,
    const unsigned short* __restrict__ Kt,
    const unsigned short* __restrict__ Vt,
    float* __restrict__ out)
{
    __shared__ unsigned short Qs[16][CC];     //  8 KB  Q tile [i][c]
    __shared__ unsigned short Vs[CC][128];    // 64 KB  V block [c][j]
    __shared__ unsigned short Ps[16][128];    //  4 KB  probs [i][j] bf16
    __shared__ float pmaxLds[8][16];
    __shared__ float psumLds[8][16];

    const int bx   = blockIdx.x;
    const int b    = bx / (NN / 16);
    const int i0   = (bx % (NN / 16)) * 16;
    const int t    = threadIdx.x;
    const int w    = t >> 5;
    const int lane = t & 31;
    const int m16  = lane & 15;
    const int h    = lane >> 4;

    {   // stage Q tile (one 32B vector per thread)
        const int m = t >> 4;
        const int c = (t & 15) * 16;
        *(v8u*)&Qs[m][c] = *(const v8u*)&Qt[((size_t)b * NN + i0 + m) * CC + c];
    }

    float rm[8], l[8], alpha[8];
    v8f oacc[2];
    #pragma unroll
    for (int r = 0; r < 8; ++r) { rm[r] = -3.0e38f; l[r] = 0.0f; }
    #pragma unroll
    for (int tile = 0; tile < 2; ++tile)
        #pragma unroll
        for (int r = 0; r < 8; ++r) oacc[tile][r] = 0.0f;

    for (int j0 = 0; j0 < NN; j0 += 128) {
        __syncthreads();   // Qs ready (1st iter); Vs/Ps/pmax safe to overwrite

        // ---- kick off async V staging: Vs[c][0..127] <- Vt[b][c][j0..] -----
        // 4096 x 16B chunks, 16 per thread; chunk = (c<<4) | j16
        {
            #pragma unroll
            for (int k = 0; k < 16; ++k) {
                const int chunk = t + k * 256;
                const int c   = chunk >> 4;
                const int j16 = (chunk & 15) * 8;       // in bf16 elements
                async_copy_b128(&Vs[c][j16],
                                &Vt[((size_t)b * CC + c) * NN + j0 + j16]);
            }
        }

        // ---- S = Q . K^T for this wave's 16 key columns -------------------
        v8f s;
        #pragma unroll
        for (int r = 0; r < 8; ++r) s[r] = 0.0f;
        const size_t krow = ((size_t)b * NN + j0 + w * 16 + m16) * CC;
        __builtin_prefetch(&Kt[krow + (size_t)128 * CC], 0, 1);  // next key block
        #pragma unroll
        for (int kc = 0; kc < CC; kc += 32) {
            ABReg a, bmat;
            a.q[0] = *(const u32x4*)&Qs[m16][kc + h * 16];
            a.q[1] = *(const u32x4*)&Qs[m16][kc + h * 16 + 8];
            bmat.w = *(const v8u*)&Kt[krow + kc + h * 16];   // L2-resident
            s = __builtin_amdgcn_wmma_f32_16x16x32_bf16(
                    false, a.v, false, bmat.v, (short)0, s, false, false);
        }

        // ---- online softmax (block-wide over 128 keys) --------------------
        float pm[8];
        #pragma unroll
        for (int r = 0; r < 8; ++r) {           // wave row-max over 16 lanes
            float v = s[r];
            v = fmaxf(v, __shfl_xor(v, 1, 32));
            v = fmaxf(v, __shfl_xor(v, 2, 32));
            v = fmaxf(v, __shfl_xor(v, 4, 32));
            v = fmaxf(v, __shfl_xor(v, 8, 32));
            pm[r] = v;
        }
        if (m16 == 0) {
            #pragma unroll
            for (int r = 0; r < 8; ++r) pmaxLds[w][r + 8 * h] = pm[r];
        }
        __syncthreads();

        #pragma unroll
        for (int r = 0; r < 8; ++r) {
            const int row = r + 8 * h;
            float bmx = pmaxLds[0][row];
            #pragma unroll
            for (int ww = 1; ww < 8; ++ww) bmx = fmaxf(bmx, pmaxLds[ww][row]);
            const float nm = fmaxf(rm[r], bmx);
            alpha[r] = __expf(rm[r] - nm);
            rm[r] = nm;
            const float p = __expf(s[r] - nm);
            s[r] = p;
            Ps[row][w * 16 + m16] = f2bf(p);
            float sum = p;                       // wave row-sum over 16 lanes
            sum += __shfl_xor(sum, 1, 32);
            sum += __shfl_xor(sum, 2, 32);
            sum += __shfl_xor(sum, 4, 32);
            sum += __shfl_xor(sum, 8, 32);
            if (m16 == 0) psumLds[w][row] = sum;
            oacc[0][r] *= alpha[r];
            oacc[1][r] *= alpha[r];
        }
        wait_async0();       // this wave's V copies are in LDS
        __syncthreads();     // all waves' Ps + Vs + psum visible

        #pragma unroll
        for (int r = 0; r < 8; ++r) {
            const int row = r + 8 * h;
            float bs = psumLds[0][row];
            #pragma unroll
            for (int ww = 1; ww < 8; ++ww) bs += psumLds[ww][row];
            l[r] = l[r] * alpha[r] + bs;
        }

        // ---- O^T[i][c-slice] += P . V^T -----------------------------------
        #pragma unroll
        for (int tile = 0; tile < 2; ++tile) {
            const int cb = w * 32 + tile * 16;
            v8f cacc = oacc[tile];
            #pragma unroll
            for (int jc = 0; jc < 128; jc += 32) {
                ABReg a, bmat;
                a.q[0] = *(const u32x4*)&Ps[m16][jc + h * 16];
                a.q[1] = *(const u32x4*)&Ps[m16][jc + h * 16 + 8];
                bmat.w = *(const v8u*)&Vs[cb + m16][jc + h * 16];
                cacc = __builtin_amdgcn_wmma_f32_16x16x32_bf16(
                           false, a.v, false, bmat.v, (short)0, cacc, false, false);
            }
            oacc[tile] = cacc;
        }
    }

    // epilogue: out[b][c][i] = O^T[i][c] / l[i]
    #pragma unroll
    for (int tile = 0; tile < 2; ++tile) {
        const int cb = w * 32 + tile * 16 + m16;
        #pragma unroll
        for (int r = 0; r < 8; ++r) {
            const int row = r + 8 * h;
            out[((size_t)b * CC + cb) * NN + i0 + row] = oacc[tile][r] / l[r];
        }
    }
}

// ---------------------------------------------------------------------------
extern "C" void kernel_launch(void* const* d_in, const int* in_sizes, int n_in,
                              void* d_out, int out_size, void* d_ws, size_t ws_size,
                              hipStream_t stream)
{
    const float* x1 = (const float*)d_in[0];
    const float* x2 = (const float*)d_in[1];
    const float* w1 = (const float*)d_in[2];
    const float* b1 = (const float*)d_in[3];
    const float* w2 = (const float*)d_in[4];
    const float* b2 = (const float*)d_in[5];
    float* out = (float*)d_out;

    unsigned short* Qt = (unsigned short*)d_ws;                 // 8 MB
    unsigned short* Kt = Qt + (size_t)BB * NN * CC;             // 8 MB
    unsigned short* Vt = Kt + (size_t)BB * NN * CC;             // 8 MB

    proj_kernel<<<BB * (NN / 16) * 2, 256, 0, stream>>>(x1, x2, w1, b1, w2, b2, Qt, Kt, Vt);
    attn_kernel<<<BB * (NN / 16), 256, 0, stream>>>(Qt, Kt, Vt, out);
}
